// ChainAwareAttention_34033320853804
// MI455X (gfx1250) — matrix operand order repaired and checked by reading, not compiled, not run on hardware
//
#include <hip/hip_runtime.h>
#include <hip/hip_bf16.h>
#include <math.h>

typedef __attribute__((ext_vector_type(16))) _Float16 v16h;
typedef __attribute__((ext_vector_type(8)))  _Float16 v8h;
typedef __attribute__((ext_vector_type(8)))  float    v8f;
typedef __attribute__((ext_vector_type(4)))  int      v4i;

#define NB 8
#define NS 512
#define ND 1024
#define NH 16
#define HD 64
#define SF (NB * NS)   // 4096 flattened (b,s) rows

#if __has_builtin(__builtin_amdgcn_global_load_async_to_lds_b128)
#define HAVE_ASYNC 1
#endif
#if __has_builtin(__builtin_amdgcn_s_wait_asynccnt)
#define HAVE_WAIT_ASYNC_BUILTIN 1
#endif

#define AS1 __attribute__((address_space(1)))
#define AS3 __attribute__((address_space(3)))

static __device__ __forceinline__ v8f wmma16(v16h a, v16h b, v8f c) {
  // D = A(16x32 f16) * B(32x16 f16) + C(16x16 f32)
  return __builtin_amdgcn_wmma_f32_16x16x32_f16(false, a, false, b, (short)0, c,
                                                false, false);
}

static __device__ __forceinline__ v16h cat8(v8h lo, v8h hi) {
  v16h t;
#pragma unroll
  for (int e = 0; e < 8; ++e) { t[e] = lo[e]; t[e + 8] = hi[e]; }
  return t;
}

static __device__ __forceinline__ v8f vzero8() {
  v8f z = {0.f, 0.f, 0.f, 0.f, 0.f, 0.f, 0.f, 0.f};
  return z;
}

static __device__ __forceinline__ void async_copy16(const _Float16* src,
                                                    _Float16* lds_dst) {
#ifdef HAVE_ASYNC
  __builtin_amdgcn_global_load_async_to_lds_b128(
      (AS1 v4i*)(v4i*)(void*)src, (AS3 v4i*)(v4i*)(void*)lds_dst, 0, 0);
#else
  *(v8h*)lds_dst = *(const v8h*)src;
#endif
}

static __device__ __forceinline__ void wait_async0() {
#if defined(HAVE_WAIT_ASYNC_BUILTIN)
  __builtin_amdgcn_s_wait_asynccnt(0);
#elif defined(HAVE_ASYNC)
  asm volatile("s_wait_asynccnt 0x0" ::: "memory");
#endif
}

// ---------------------------------------------------------------------------
// Pack x (f32, [4096 x 1024]) into A-fragment-ordered f16 tiles.
// A-frag mapping: lane L: row = L&15; element e: K = (L>>4)*16 + e
// ---------------------------------------------------------------------------
__global__ __launch_bounds__(32) void pack_x_kernel(const float* __restrict__ X,
                                                    _Float16* __restrict__ Xp) {
  int kt = blockIdx.x;   // 0..31
  int st = blockIdx.y;   // 0..255
  int L = threadIdx.x;
  int m = L & 15, hh = L >> 4;
  const float* src = X + (size_t)(st * 16 + m) * ND + kt * 32;
  _Float16 vals[16];
#pragma unroll
  for (int e = 0; e < 16; ++e) {
    int k = hh * 16 + e;
    vals[e] = (_Float16)src[k];
  }
  _Float16* dst = Xp + ((size_t)(st * 32 + kt) * 512) + L * 16;
  *(v16h*)dst = *(const v16h*)vals;
}

// ---------------------------------------------------------------------------
// Pack a [1024 x 1024] f32 weight into B-fragment-ordered f16 tiles.
// B-frag mapping: lane L: col = L&15; element e: K = (L>>4)*16 + e
// ---------------------------------------------------------------------------
__global__ __launch_bounds__(32) void pack_w_kernel(const float* __restrict__ W,
                                                    _Float16* __restrict__ Wp) {
  int kt = blockIdx.x;   // 0..31
  int ft = blockIdx.y;   // 0..63
  int L = threadIdx.x;
  int n = L & 15, hh = L >> 4;
  _Float16 vals[16];
#pragma unroll
  for (int e = 0; e < 16; ++e) {
    int k = kt * 32 + hh * 16 + e;
    vals[e] = (_Float16)W[(size_t)k * ND + ft * 16 + n];
  }
  _Float16* dst = Wp + ((size_t)(ft * 32 + kt) * 512) + L * 16;
  *(v16h*)dst = *(const v16h*)vals;
}

// ---------------------------------------------------------------------------
// Projection GEMM v2: 4-wave workgroup computes a 64x128 tile.
// Wave (wy,wx): rows mb*64 + wy*32 (+2 subtiles), cols nb*128 + wx*64.
// A(4 tiles) + B(8 tiles) staged per K-step in double-buffered LDS via
// GLOBAL_LOAD_ASYNC_TO_LDS_B128 (ASYNCcnt pipeline), 8 wmma per wave per step.
// z selects weight: 0:Qs 1:Ks 2:Vs(T) 3:Qc 4:Kc 5:Vc(T)
// ---------------------------------------------------------------------------
__global__ __launch_bounds__(128) void proj_gemm_kernel(
    const _Float16* __restrict__ Xp, const _Float16* __restrict__ Wp6,
    _Float16* __restrict__ Qs, _Float16* __restrict__ Ks,
    _Float16* __restrict__ Vst, _Float16* __restrict__ Qc,
    _Float16* __restrict__ Kc, _Float16* __restrict__ Vct) {
  int mb = blockIdx.x;   // 0..63   (64-row block)
  int nb = blockIdx.y;   // 0..7    (128-col block)
  int w  = blockIdx.z;   // 0..5
  int tid = threadIdx.x;
  int L = tid & 31;
  int wave = tid >> 5;   // 0..3
  int wy = wave >> 1;    // row half
  int wx = wave & 1;     // col half
  const _Float16* Wp = Wp6 + (size_t)w * (ND * ND);

  // [buf][12 tiles x 512 f16]: segs 0..3 = A tiles, segs 4..11 = B tiles
  __shared__ __align__(16) _Float16 sbuf[2][12 * 512];

  auto stage = [&](int kt, int pbuf) {
#pragma unroll
    for (int i = 0; i < 6; ++i) {
      int idx = tid + i * 128;          // 0..767 (16B transfers)
      int seg = idx >> 6;               // 64 transfers per 1KB segment
      int off = (idx & 63) * 8;         // f16 offset
      const _Float16* src =
          (seg < 4)
              ? Xp + ((size_t)((mb * 4 + seg) * 32 + kt) * 512) + off
              : Wp + ((size_t)((nb * 8 + (seg - 4)) * 32 + kt) * 512) + off;
      async_copy16(src, &sbuf[pbuf][seg * 512 + off]);
    }
  };

  v8f acc[8];
#pragma unroll
  for (int i = 0; i < 8; ++i) acc[i] = vzero8();

  stage(0, 0);
  for (int kt = 0; kt < 32; ++kt) {
    int pb = kt & 1;
    wait_async0();        // own async stores for sbuf[pb] complete
    __syncthreads();      // all waves' stage-kt data visible; prior reads done
    if (kt + 1 < 32) stage(kt + 1, pb ^ 1);   // overlap next stage w/ compute

    const _Float16* base = sbuf[pb];
    v16h a0 = *(const v16h*)(base + (size_t)(wy * 2 + 0) * 512 + L * 16);
    v16h a1 = *(const v16h*)(base + (size_t)(wy * 2 + 1) * 512 + L * 16);
#pragma unroll
    for (int fi = 0; fi < 4; ++fi) {
      v16h bf = *(const v16h*)(base + (size_t)(4 + wx * 4 + fi) * 512 + L * 16);
      acc[fi]     = wmma16(a0, bf, acc[fi]);
      acc[4 + fi] = wmma16(a1, bf, acc[4 + fi]);
    }
  }

  int n = L & 15, hh = L >> 4;
#pragma unroll
  for (int sub = 0; sub < 2; ++sub) {
#pragma unroll
    for (int fi = 0; fi < 4; ++fi) {
#pragma unroll
      for (int r = 0; r < 8; ++r) {
        int sflat = mb * 64 + wy * 32 + sub * 16 + hh * 8 + r;  // C-layout row
        int b = sflat >> 9, s = sflat & (NS - 1);
        int f = nb * 128 + wx * 64 + fi * 16 + n;
        int head = f >> 6, d = f & (HD - 1);
        _Float16 hv = (_Float16)acc[sub * 4 + fi][r];
        size_t bh = (size_t)(b * NH + head);
        if (w == 0)      Qs [(bh * NS + s) * HD + d] = hv;
        else if (w == 1) Ks [(bh * NS + s) * HD + d] = hv;
        else if (w == 2) Vst[(bh * HD + d) * NS + s] = hv;
        else if (w == 3) Qc [(bh * NS + s) * HD + d] = hv;
        else if (w == 4) Kc [(bh * NS + s) * HD + d] = hv;
        else             Vct[(bh * HD + d) * NS + s] = hv;
      }
    }
  }
}

// ---------------------------------------------------------------------------
// RoPE on Qs, Ks in place. One thread per (bh, s, dpair); handles d and d+32.
// ---------------------------------------------------------------------------
__global__ void rope_kernel(_Float16* __restrict__ Qs, _Float16* __restrict__ Ks) {
  size_t idx = (size_t)blockIdx.x * blockDim.x + threadIdx.x;
  if (idx >= (size_t)NB * NH * NS * 32) return;
  int dp = (int)(idx & 31);
  int s  = (int)((idx >> 5) & (NS - 1));
  size_t bh = idx >> 14;
  float invf = __expf(-(float)dp * (9.210340371976184f / 32.0f)); // 10000^(-dp/32)
  float fr = (float)s * invf;
  float c = __cosf(fr), sn = __sinf(fr);
  size_t base = (bh * NS + s) * HD;
  float q1 = (float)Qs[base + dp], q2 = (float)Qs[base + dp + 32];
  Qs[base + dp]      = (_Float16)(q1 * c - q2 * sn);
  Qs[base + dp + 32] = (_Float16)(q2 * c + q1 * sn);
  float k1 = (float)Ks[base + dp], k2 = (float)Ks[base + dp + 32];
  Ks[base + dp]      = (_Float16)(k1 * c - k2 * sn);
  Ks[base + dp + 32] = (_Float16)(k2 * c + k1 * sn);
}

// ---------------------------------------------------------------------------
// Chain-aware flash attention. One wave per (b, head, 16-query tile).
// ---------------------------------------------------------------------------
__global__ __launch_bounds__(32) void attn_kernel(
    const _Float16* __restrict__ Qs, const _Float16* __restrict__ Ks,
    const _Float16* __restrict__ Vst, const _Float16* __restrict__ Qc,
    const _Float16* __restrict__ Kc, const _Float16* __restrict__ Vct,
    const int* __restrict__ cid, const unsigned char* __restrict__ am,
    _Float16* __restrict__ AO) {
  int qt = blockIdx.x, head = blockIdx.y, b = blockIdx.z;
  int L = threadIdx.x;
  int n = L & 15, hh = L >> 4, m = n;   // m: A-frag row, n: B/C col
  size_t bh = (size_t)(b * NH + head);

  __shared__ __align__(16) _Float16 pS[16 * 32];
  __shared__ __align__(16) _Float16 pC[16 * 32];

  const _Float16* qsrow = Qs + (bh * NS + qt * 16 + m) * HD;
  const _Float16* qcrow = Qc + (bh * NS + qt * 16 + m) * HD;
  v16h aQs[2], aQc[2];
#pragma unroll
  for (int f = 0; f < 2; ++f) {
    aQs[f] = cat8(*(const v8h*)(qsrow + f * 32 + hh * 16),
                  *(const v8h*)(qsrow + f * 32 + hh * 16 + 8));
    aQc[f] = cat8(*(const v8h*)(qcrow + f * 32 + hh * 16),
                  *(const v8h*)(qcrow + f * 32 + hh * 16 + 8));
  }
  int cidq[8];
#pragma unroll
  for (int r = 0; r < 8; ++r) cidq[r] = cid[b * NS + qt * 16 + hh * 8 + r];

  float m_run[8], l_run[8];
#pragma unroll
  for (int r = 0; r < 8; ++r) { m_run[r] = -1e30f; l_run[r] = 0.f; }
  v8f accS[4], accC[4];
#pragma unroll
  for (int dt = 0; dt < 4; ++dt) { accS[dt] = vzero8(); accC[dt] = vzero8(); }

  for (int ck = 0; ck < 16; ++ck) {
    float mg[2][8];
    int cidk[2], mk[2];
#pragma unroll
    for (int t = 0; t < 2; ++t) {
      int k16 = ck * 2 + t;
      const _Float16* ksrow = Ks + (bh * NS + k16 * 16 + n) * HD;
      const _Float16* kcrow = Kc + (bh * NS + k16 * 16 + n) * HD;
      v16h kb0s = *(const v16h*)(ksrow + hh * 16);
      v16h kb1s = *(const v16h*)(ksrow + 32 + hh * 16);
      v16h kb0c = *(const v16h*)(kcrow + hh * 16);
      v16h kb1c = *(const v16h*)(kcrow + 32 + hh * 16);
      v8f ss = vzero8();
      ss = wmma16(aQs[0], kb0s, ss);
      ss = wmma16(aQs[1], kb1s, ss);
      v8f sc = vzero8();
      sc = wmma16(aQc[0], kb0c, sc);
      sc = wmma16(aQc[1], kb1c, sc);
      int key = k16 * 16 + n;
      cidk[t] = cid[b * NS + key];
      mk[t] = am[b * NS + key];
#pragma unroll
      for (int r = 0; r < 8; ++r) {
        float v = (cidq[r] == cidk[t]) ? ss[r] * 0.125f : sc[r] * 0.125f;
        if (!mk[t]) v = -INFINITY;
        mg[t][r] = v;
      }
    }
    float scl[8], pbuf[2][8];
#pragma unroll
    for (int r = 0; r < 8; ++r) {
      float rm = fmaxf(mg[0][r], mg[1][r]);
#pragma unroll
      for (int off = 1; off < 16; off <<= 1) rm = fmaxf(rm, __shfl_xor(rm, off));
      float mnew = fmaxf(m_run[r], rm);
      float sc = __expf(m_run[r] - mnew);
      float p0 = __expf(mg[0][r] - mnew);
      float p1 = __expf(mg[1][r] - mnew);
      float rs = p0 + p1;
#pragma unroll
      for (int off = 1; off < 16; off <<= 1) rs += __shfl_xor(rs, off);
      m_run[r] = mnew;
      l_run[r] = l_run[r] * sc + rs;
      scl[r] = sc;
      pbuf[0][r] = p0;
      pbuf[1][r] = p1;
    }
#pragma unroll
    for (int dt = 0; dt < 4; ++dt)
#pragma unroll
      for (int r = 0; r < 8; ++r) { accS[dt][r] *= scl[r]; accC[dt][r] *= scl[r]; }

#pragma unroll
    for (int t = 0; t < 2; ++t)
#pragma unroll
      for (int r = 0; r < 8; ++r) {
        bool intra = (cidq[r] == cidk[t]);
        float p = pbuf[t][r];
        pS[(hh * 8 + r) * 32 + t * 16 + n] = (_Float16)(intra ? p : 0.f);
        pC[(hh * 8 + r) * 32 + t * 16 + n] = (_Float16)(intra ? 0.f : p);
      }
    __syncthreads();
    v16h pAs = cat8(*(const v8h*)(pS + m * 32 + hh * 16),
                    *(const v8h*)(pS + m * 32 + hh * 16 + 8));
    v16h pAc = cat8(*(const v8h*)(pC + m * 32 + hh * 16),
                    *(const v8h*)(pC + m * 32 + hh * 16 + 8));
#pragma unroll
    for (int dt = 0; dt < 4; ++dt) {
      v16h vs = *(const v16h*)(Vst + (bh * HD + dt * 16 + n) * NS + ck * 32 + hh * 16);
      v16h vc = *(const v16h*)(Vct + (bh * HD + dt * 16 + n) * NS + ck * 32 + hh * 16);
      accS[dt] = wmma16(pAs, vs, accS[dt]);
      accC[dt] = wmma16(pAc, vc, accC[dt]);
    }
    __syncthreads();
  }

#pragma unroll
  for (int r = 0; r < 8; ++r) {
    float rl = (l_run[r] > 0.f) ? 1.f / l_run[r] : 0.f;
    int q = qt * 16 + hh * 8 + r;
#pragma unroll
    for (int dt = 0; dt < 4; ++dt) {
      float v = (accS[dt][r] + accC[dt][r]) * rl;
      AO[(bh * NS + q) * HD + dt * 16 + n] = (_Float16)v;
    }
  }
}

// ---------------------------------------------------------------------------
// Output projection v2: 32 rows x 64 cols per wave (8 wmma per K-step).
// ---------------------------------------------------------------------------
__global__ __launch_bounds__(32) void final_gemm_kernel(
    const _Float16* __restrict__ AO, const _Float16* __restrict__ Wop,
    float* __restrict__ Out) {
  int st = blockIdx.x;   // 0..127 (32-row blocks)
  int fg = blockIdx.y;   // 0..15  (64-col blocks)
  int L = threadIdx.x;
  int m = L & 15, hh = L >> 4, n = m;

  v8f acc[8];
#pragma unroll
  for (int i = 0; i < 8; ++i) acc[i] = vzero8();

  for (int kt = 0; kt < 32; ++kt) {
    int head = kt >> 1;
    int db = (kt & 1) * 32;
    v16h a[2];
#pragma unroll
    for (int sub = 0; sub < 2; ++sub) {
      int sflat = st * 32 + sub * 16 + m;
      int b = sflat >> 9, s = sflat & (NS - 1);
      const _Float16* arow =
          AO + (((size_t)(b * NH + head) * NS) + s) * HD + db;
      a[sub] = cat8(*(const v8h*)(arow + hh * 16),
                    *(const v8h*)(arow + hh * 16 + 8));
    }
#pragma unroll
    for (int fi = 0; fi < 4; ++fi) {
      int ft = fg * 4 + fi;
      v16h bf = *(const v16h*)(Wop + ((size_t)(ft * 32 + kt) * 512) + L * 16);
      acc[fi]     = wmma16(a[0], bf, acc[fi]);
      acc[4 + fi] = wmma16(a[1], bf, acc[4 + fi]);
    }
  }
#pragma unroll
  for (int sub = 0; sub < 2; ++sub) {
#pragma unroll
    for (int fi = 0; fi < 4; ++fi) {
#pragma unroll
      for (int r = 0; r < 8; ++r) {
        int so = st * 32 + sub * 16 + hh * 8 + r;      // C-layout row
        int f = fg * 64 + fi * 16 + n;
        Out[(size_t)so * ND + f] = acc[sub * 4 + fi][r];
      }
    }
  }
}

// ---------------------------------------------------------------------------
extern "C" void kernel_launch(void* const* d_in, const int* in_sizes, int n_in,
                              void* d_out, int out_size, void* d_ws, size_t ws_size,
                              hipStream_t stream) {
  (void)in_sizes; (void)n_in; (void)out_size; (void)ws_size;
  const float* x            = (const float*)d_in[0];
  const int* chain_ids      = (const int*)d_in[1];
  const unsigned char* mask = (const unsigned char*)d_in[2];
  const float* Wmats[6] = {
      (const float*)d_in[3],  // Wq_self
      (const float*)d_in[4],  // Wk_self
      (const float*)d_in[5],  // Wv_self
      (const float*)d_in[6],  // Wq_cross
      (const float*)d_in[7],  // Wk_cross
      (const float*)d_in[8]   // Wv_cross
  };
  const float* Wo = (const float*)d_in[9];
  float* out = (float*)d_out;

  _Float16* ws = (_Float16*)d_ws;
  size_t o = 0;
  const size_t NQK = (size_t)NB * NH * NS * HD;   // 4,194,304 f16 per tensor
  _Float16* Xp  = ws + o; o += (size_t)SF * ND;
  _Float16* Wp6 = ws + o; o += (size_t)6 * ND * ND;
  _Float16* Wop = ws + o; o += (size_t)ND * ND;
  _Float16* Qs  = ws + o; o += NQK;
  _Float16* Ks  = ws + o; o += NQK;
  _Float16* Vst = ws + o; o += NQK;
  _Float16* Qc  = ws + o; o += NQK;
  _Float16* Kc  = ws + o; o += NQK;
  _Float16* Vct = ws + o; o += NQK;
  _Float16* AO  = ws + o; o += NQK;

  pack_x_kernel<<<dim3(32, 256), 32, 0, stream>>>(x, Xp);
  for (int i = 0; i < 6; ++i)
    pack_w_kernel<<<dim3(32, 64), 32, 0, stream>>>(Wmats[i], Wp6 + (size_t)i * ND * ND);
  pack_w_kernel<<<dim3(32, 64), 32, 0, stream>>>(Wo, Wop);

  proj_gemm_kernel<<<dim3(64, 8, 6), 128, 0, stream>>>(Xp, Wp6, Qs, Ks, Vst, Qc, Kc, Vct);

  {
    size_t total = (size_t)NB * NH * NS * 32;
    int threads = 256;
    int blocks = (int)((total + threads - 1) / threads);
    rope_kernel<<<blocks, threads, 0, stream>>>(Qs, Ks);
  }

  attn_kernel<<<dim3(NS / 16, NH, NB), 32, 0, stream>>>(Qs, Ks, Vst, Qc, Kc, Vct,
                                                        chain_ids, mask, AO);

  final_gemm_kernel<<<dim3(128, 16), 32, 0, stream>>>(AO, Wop, out);
}
